// Transformer_46651934769215
// MI455X (gfx1250) — compile-verified, matrix operand order checked by reading, not run on hardware
//
#include <hip/hip_runtime.h>
#include <hip/hip_bf16.h>
#include <math.h>

// ---------------------------------------------------------------------------
// Transformer encoder forward for MI455X (gfx1250), all GEMMs on
// v_wmma_f32_16x16x32_f16 (wave32). Flash-style attention (no SxS buffer),
// async global->LDS staging for K tiles, 32x64 wave tiles in the FFN GEMMs.
// ---------------------------------------------------------------------------

typedef _Float16 h16;
typedef __attribute__((ext_vector_type(16))) _Float16 v16h;
typedef __attribute__((ext_vector_type(8)))  _Float16 v8h;
typedef __attribute__((ext_vector_type(8)))  float    v8f;

#define B_  32
#define S_  512
#define E_  256
#define H_  8
#define DH_ 32
#define L_  4
#define FF_ 1024
#define EPS_ 1e-5f

// ---- WMMA helpers ---------------------------------------------------------

__device__ __forceinline__ v8f wmma_f16(v16h a, v16h b, v8f c) {
  // D = A(16x32 f16) * B(32x16 f16) + C(16x16 f32)
  return __builtin_amdgcn_wmma_f32_16x16x32_f16(
      /*neg_a=*/false, a, /*neg_b=*/false, b,
      /*c_mod=*/(short)0, c, /*reuse_a=*/false, /*reuse_b=*/false);
}

// Load an A fragment (16x32, row-major, leading dim `ld` in halves), or a
// B fragment whose memory image is B^T row-major (i.e. mem[n][k] contiguous
// in k). Per ISA 7.12.2: lane L: row = L&15; half j holds
// K = (j&7) + 16*(j>>3) + 8*(L>>4)  -> two contiguous 8-half (16B) chunks.
__device__ __forceinline__ v16h frag_ld(const h16* base, int ld) {
  const int lane = threadIdx.x & 31;
  const int r = lane & 15, hi = lane >> 4;
  const h16* p = base + r * ld + 8 * hi;
  v8h a = *reinterpret_cast<const v8h*>(p);
  v8h b = *reinterpret_cast<const v8h*>(p + 16);
  v16h o;
#pragma unroll
  for (int i = 0; i < 8; ++i) { o[i] = a[i]; o[8 + i] = b[i]; }
  return o;
}

// Async global->LDS copy, 16B per lane (GLOBAL_LOAD_ASYNC_TO_LDS_B128, GV
// mode; LDS byte offset is the low 32 bits of the flat shared address).
__device__ __forceinline__ void async_copy_b128(unsigned lds_off, const void* gaddr) {
  asm volatile("global_load_async_to_lds_b128 %0, %1, off"
               :: "v"(lds_off), "v"(gaddr) : "memory");
}
__device__ __forceinline__ void async_wait0() {
  asm volatile("s_wait_asynccnt 0x0" ::: "memory");
}

// ---- Elementwise / prep kernels -------------------------------------------

__global__ void k_addpos(const float* __restrict__ x,
                         const float* __restrict__ pos,
                         float* __restrict__ h) {
  int i = blockIdx.x * 256 + threadIdx.x;          // B*S*E total
  h[i] = x[i] + pos[i & (E_ - 1)];
}

// Convert fp32 [nmat][R][C] -> f16 [nmat][C][R] (transpose last two dims).
__global__ void k_convT(const float* __restrict__ in, h16* __restrict__ out,
                        int R, int C) {
  int i = blockIdx.x * 256 + threadIdx.x;
  int rc = R * C;
  int m = i / rc, rem = i - m * rc;
  int r = rem / C, c = rem - r * C;
  out[m * rc + c * R + r] = (h16)in[i];
}

// LayerNorm over E=256 (one row per 256-thread block), output f16.
__global__ void k_ln(const float* __restrict__ x, h16* __restrict__ y) {
  __shared__ float red[256];
  int row = blockIdx.x, t = threadIdx.x;
  float v = x[row * E_ + t];
  red[t] = v;
  __syncthreads();
  for (int s = 128; s > 0; s >>= 1) { if (t < s) red[t] += red[t + s]; __syncthreads(); }
  float mean = red[0] * (1.f / E_);
  __syncthreads();
  float d = v - mean;
  red[t] = d * d;
  __syncthreads();
  for (int s = 128; s > 0; s >>= 1) { if (t < s) red[t] += red[t + s]; __syncthreads(); }
  float var = red[0] * (1.f / E_);
  y[row * E_ + t] = (h16)(d * rsqrtf(var + EPS_));
}

// res = mha + h ; rn = f16(LayerNorm(res))
__global__ void k_resln(const float* __restrict__ mha, const float* __restrict__ h,
                        float* __restrict__ res, h16* __restrict__ rn) {
  __shared__ float red[256];
  int row = blockIdx.x, t = threadIdx.x;
  float v = mha[row * E_ + t] + h[row * E_ + t];
  res[row * E_ + t] = v;
  red[t] = v;
  __syncthreads();
  for (int s = 128; s > 0; s >>= 1) { if (t < s) red[t] += red[t + s]; __syncthreads(); }
  float mean = red[0] * (1.f / E_);
  __syncthreads();
  float d = v - mean;
  red[t] = d * d;
  __syncthreads();
  for (int s = 128; s > 0; s >>= 1) { if (t < s) red[t] += red[t + s]; __syncthreads(); }
  float var = red[0] * (1.f / E_);
  rn[row * E_ + t] = (h16)(d * rsqrtf(var + EPS_));
}

// ---- QKV projection: per (mat,b,h) S x DH = (S x 32)*(32 x 32) -----------
// One wave = one 16x16 output tile, single K=32 WMMA.
__global__ void __launch_bounds__(256) k_qkv(
    const h16* __restrict__ xn,
    const h16* __restrict__ WqT, const h16* __restrict__ WkT,
    const h16* __restrict__ WvT,
    h16* __restrict__ q, h16* __restrict__ k, h16* __restrict__ v, int layer) {
  int wave = (blockIdx.x * 256 + threadIdx.x) >> 5;
  int nt = wave & 1;  wave >>= 1;   // DH tile (2)
  int mt = wave & 31; wave >>= 5;   // S tile (32)
  int hh = wave & 7;  wave >>= 3;   // head
  int b  = wave & 31; wave >>= 5;   // batch
  int mat = wave;                   // 0=q 1=k 2=v
  const h16* WT = (mat == 0) ? WqT : (mat == 1) ? WkT : WvT;
  h16* dst = (mat == 0) ? q : (mat == 1) ? k : v;

  v16h a  = frag_ld(xn + (b * S_ + mt * 16) * E_ + hh * DH_, E_);
  v16h bb = frag_ld(WT + ((layer * H_ + hh) * DH_ + nt * 16) * DH_, DH_);
  v8f c = {};
  c = wmma_f16(a, bb, c);

  int lane = threadIdx.x & 31;
  int n = lane & 15, hi = lane >> 4;
#pragma unroll
  for (int r = 0; r < 8; ++r) {
    int m = r + 8 * hi;
    dst[((b * H_ + hh) * S_ + mt * 16 + m) * DH_ + nt * 16 + n] = (h16)c[r];
  }
}

// ---- Flash attention: block = 4 waves, 64 query rows of one (b,h) --------
__global__ void __launch_bounds__(128) k_attn(
    const h16* __restrict__ q, const h16* __restrict__ k,
    const h16* __restrict__ v, float* __restrict__ mha) {
  __shared__ __align__(16) h16 ldsK[32 * 32];       // [key][d]
  __shared__ __align__(16) h16 ldsV[32 * 32];       // transposed: [d][key]
  __shared__ __align__(16) h16 ldsP[4 * 16 * 32];   // per-wave P tile

  int tid = threadIdx.x, wave = tid >> 5, lane = tid & 31;
  int blk = blockIdx.x;
  int qb = blk & 7;         // S/64 = 8 query blocks
  int bh = blk >> 3;        // b*H + h
  int q0 = qb * 64 + wave * 16;

  const h16* qbase = q + (bh * S_ + q0) * DH_;
  const h16* kbase = k + bh * S_ * DH_;
  const h16* vbase = v + bh * S_ * DH_;

  v16h qa = frag_ld(qbase, DH_);
  v8f acc0 = {}, acc1 = {};
  float mrow[8], srow[8];
#pragma unroll
  for (int r = 0; r < 8; ++r) { mrow[r] = -INFINITY; srow[r] = 0.f; }
  const float scl = 0.17677669529663688f;  // 1/sqrt(DH)
  h16* pw = ldsP + wave * 512;
  const unsigned ldsK_off = (unsigned)(uintptr_t)(&ldsK[0]) + (unsigned)tid * 16u;

  for (int c0 = 0; c0 < S_; c0 += 32) {
    __syncthreads();
    {  // stage K chunk async (2KB = 128 lanes x 16B), V chunk transposed
      async_copy_b128(ldsK_off, (const char*)(kbase + c0 * DH_) + tid * 16);
      const h16* vs = vbase + c0 * DH_;
      for (int i = tid; i < 1024; i += 128) {
        int rr = i >> 5, cc = i & 31;
        ldsV[cc * 32 + rr] = vs[i];
      }
      async_wait0();
    }
    __syncthreads();

    v16h bk0 = frag_ld(ldsK, DH_);
    v16h bk1 = frag_ld(ldsK + 16 * DH_, DH_);
    v8f s0 = {}, s1 = {};
    s0 = wmma_f16(qa, bk0, s0);
    s1 = wmma_f16(qa, bk1, s1);

    // online softmax update (rows r+8*hi per lane; reduce across 16 lanes)
#pragma unroll
    for (int r = 0; r < 8; ++r) {
      float a0 = s0[r] * scl, a1 = s1[r] * scl;
      float lm = fmaxf(a0, a1);
      lm = fmaxf(lm, __shfl_xor(lm, 1, 32));
      lm = fmaxf(lm, __shfl_xor(lm, 2, 32));
      lm = fmaxf(lm, __shfl_xor(lm, 4, 32));
      lm = fmaxf(lm, __shfl_xor(lm, 8, 32));
      float nm = fmaxf(mrow[r], lm);
      float p0 = __expf(a0 - nm), p1 = __expf(a1 - nm);
      int m = r + 8 * (lane >> 4);
      pw[m * 32 + (lane & 15)] = (h16)p0;
      pw[m * 32 + 16 + (lane & 15)] = (h16)p1;
      float ps = p0 + p1;
      ps += __shfl_xor(ps, 1, 32);
      ps += __shfl_xor(ps, 2, 32);
      ps += __shfl_xor(ps, 4, 32);
      ps += __shfl_xor(ps, 8, 32);
      float alpha = __expf(mrow[r] - nm);
      srow[r] = srow[r] * alpha + ps;
      mrow[r] = nm;
      acc0[r] *= alpha;
      acc1[r] *= alpha;
    }
    __syncthreads();

    v16h pa  = frag_ld(pw, 32);
    v16h bv0 = frag_ld(ldsV, 32);            // d = 0..15
    v16h bv1 = frag_ld(ldsV + 16 * 32, 32);  // d = 16..31
    acc0 = wmma_f16(pa, bv0, acc0);
    acc1 = wmma_f16(pa, bv1, acc1);
  }

  int b = bh >> 3, hh = bh & 7;
  int n = lane & 15, hi = lane >> 4;
#pragma unroll
  for (int r = 0; r < 8; ++r) {
    int m = r + 8 * hi;
    float inv = 1.f / srow[r];
    float* o = mha + (b * S_ + q0 + m) * E_ + hh * DH_;
    o[n] = acc0[r] * inv;
    o[16 + n] = acc1[r] * inv;
  }
}

// ---- FFN1: (B*S x E) * (E x FF) + b1, exact GELU, f16 out ----------------
// One wave computes a 32x64 tile: 2 A fragments, B reused across both.
__global__ void __launch_bounds__(128) k_ffn1(
    const h16* __restrict__ rn, const h16* __restrict__ W1T,
    const float* __restrict__ b1, h16* __restrict__ out, int layer) {
  int wave = (blockIdx.x * 128 + threadIdx.x) >> 5;
  int nb = wave & 15, mt = wave >> 4;  // 16 N-blocks of 64, 512 M-tiles of 32
  int row0 = mt * 32, n0 = nb * 64;
  const h16* a_base = rn + row0 * E_;
  const h16* b_base = W1T + layer * FF_ * E_;
  v8f acc[2][4] = {};
  for (int kk = 0; kk < E_; kk += 32) {
    if (kk + 32 < E_) __builtin_prefetch(a_base + kk + 32, 0, 1);
    v16h a0 = frag_ld(a_base + kk, E_);
    v16h a1 = frag_ld(a_base + 16 * E_ + kk, E_);
#pragma unroll
    for (int t = 0; t < 4; ++t) {
      v16h bb = frag_ld(b_base + (n0 + t * 16) * E_ + kk, E_);
      acc[0][t] = wmma_f16(a0, bb, acc[0][t]);
      acc[1][t] = wmma_f16(a1, bb, acc[1][t]);
    }
  }
  int lane = threadIdx.x & 31, n = lane & 15, hi = lane >> 4;
#pragma unroll
  for (int u = 0; u < 2; ++u) {
#pragma unroll
    for (int t = 0; t < 4; ++t) {
      int col = n0 + t * 16 + n;
      float bias = b1[layer * FF_ + col];
#pragma unroll
      for (int r = 0; r < 8; ++r) {
        int m = u * 16 + r + 8 * hi;
        float xg = acc[u][t][r] + bias;
        float g = 0.5f * xg * (1.f + erff(xg * 0.70710678118654752f));
        out[(row0 + m) * FF_ + col] = (h16)g;
      }
    }
  }
}

// ---- FFN2: (B*S x FF) * (FF x E) + b2 + res -> fp32 out ------------------
__global__ void __launch_bounds__(128) k_ffn2(
    const h16* __restrict__ x, const h16* __restrict__ W2T,
    const float* __restrict__ b2, const float* __restrict__ res,
    float* __restrict__ out, int layer) {
  int wave = (blockIdx.x * 128 + threadIdx.x) >> 5;
  int nb = wave & 3, mt = wave >> 2;  // 4 N-blocks of 64, 512 M-tiles of 32
  int row0 = mt * 32, n0 = nb * 64;
  const h16* a_base = x + row0 * FF_;
  const h16* b_base = W2T + layer * E_ * FF_;
  v8f acc[2][4] = {};
  for (int kk = 0; kk < FF_; kk += 32) {
    if (kk + 32 < FF_) __builtin_prefetch(a_base + kk + 32, 0, 1);
    v16h a0 = frag_ld(a_base + kk, FF_);
    v16h a1 = frag_ld(a_base + 16 * FF_ + kk, FF_);
#pragma unroll
    for (int t = 0; t < 4; ++t) {
      v16h bb = frag_ld(b_base + (n0 + t * 16) * FF_ + kk, FF_);
      acc[0][t] = wmma_f16(a0, bb, acc[0][t]);
      acc[1][t] = wmma_f16(a1, bb, acc[1][t]);
    }
  }
  int lane = threadIdx.x & 31, n = lane & 15, hi = lane >> 4;
#pragma unroll
  for (int u = 0; u < 2; ++u) {
#pragma unroll
    for (int t = 0; t < 4; ++t) {
      int col = n0 + t * 16 + n;
      float bias = b2[layer * E_ + col];
#pragma unroll
      for (int r = 0; r < 8; ++r) {
        int m = u * 16 + r + 8 * hi;
        out[(row0 + m) * E_ + col] = acc[u][t][r] + bias + res[(row0 + m) * E_ + col];
      }
    }
  }
}

// ---------------------------------------------------------------------------

extern "C" void kernel_launch(void* const* d_in, const int* in_sizes, int n_in,
                              void* d_out, int out_size, void* d_ws, size_t ws_size,
                              hipStream_t stream) {
  (void)in_sizes; (void)n_in; (void)out_size; (void)ws_size;
  const float* x   = (const float*)d_in[0];
  const float* pos = (const float*)d_in[1];
  const float* Wq  = (const float*)d_in[2];
  const float* Wk  = (const float*)d_in[3];
  const float* Wv  = (const float*)d_in[4];
  const float* W1  = (const float*)d_in[5];
  const float* b1  = (const float*)d_in[6];
  const float* W2  = (const float*)d_in[7];
  const float* b2  = (const float*)d_in[8];
  float* outp = (float*)d_out;

  char* w = (char*)d_ws;
  size_t off = 0;
  const size_t BSE = (size_t)B_ * S_ * E_;
  float* h    = (float*)(w + off); off += BSE * 4;
  h16* xn16   = (h16*)(w + off);   off += BSE * 2;
  h16* q16    = (h16*)(w + off);   off += BSE * 2;
  h16* k16    = (h16*)(w + off);   off += BSE * 2;
  h16* v16    = (h16*)(w + off);   off += BSE * 2;
  float* mha  = (float*)(w + off); off += BSE * 4;
  float* res  = (float*)(w + off); off += BSE * 4;
  h16* rn16   = (h16*)(w + off);   off += BSE * 2;
  h16* ffn16  = (h16*)(w + off);   off += (size_t)B_ * S_ * FF_ * 2;
  h16* WqT    = (h16*)(w + off);   off += (size_t)L_ * H_ * DH_ * DH_ * 2;
  h16* WkT    = (h16*)(w + off);   off += (size_t)L_ * H_ * DH_ * DH_ * 2;
  h16* WvT    = (h16*)(w + off);   off += (size_t)L_ * H_ * DH_ * DH_ * 2;
  h16* W1T    = (h16*)(w + off);   off += (size_t)L_ * E_ * FF_ * 2;
  h16* W2T    = (h16*)(w + off);   off += (size_t)L_ * FF_ * E_ * 2;

  // Weight convert + transpose (deterministic each call).
  k_convT<<<(L_ * H_ * DH_ * DH_) / 256, 256, 0, stream>>>(Wq, WqT, DH_, DH_);
  k_convT<<<(L_ * H_ * DH_ * DH_) / 256, 256, 0, stream>>>(Wk, WkT, DH_, DH_);
  k_convT<<<(L_ * H_ * DH_ * DH_) / 256, 256, 0, stream>>>(Wv, WvT, DH_, DH_);
  k_convT<<<(L_ * E_ * FF_) / 256, 256, 0, stream>>>(W1, W1T, E_, FF_);
  k_convT<<<(L_ * FF_ * E_) / 256, 256, 0, stream>>>(W2, W2T, FF_, E_);

  k_addpos<<<(int)(BSE / 256), 256, 0, stream>>>(x, pos, h);

  const int rows = B_ * S_;  // 16384
  for (int l = 0; l < L_; ++l) {
    k_ln<<<rows, 256, 0, stream>>>(h, xn16);
    k_qkv<<<6144, 256, 0, stream>>>(xn16, WqT, WkT, WvT, q16, k16, v16, l);
    k_attn<<<B_ * H_ * (S_ / 64), 128, 0, stream>>>(q16, k16, v16, mha);
    k_resln<<<rows, 256, 0, stream>>>(mha, h, res, rn16);
    k_ffn1<<<2048, 128, 0, stream>>>(rn16, W1T, b1, ffn16, l);
    float* dst = (l == L_ - 1) ? outp : h;
    k_ffn2<<<512, 128, 0, stream>>>(ffn16, W2T, b2, res, dst, l);
  }
}
